// ScaledDotProduct_46213848104965
// MI455X (gfx1250) — matrix operand order repaired and checked by reading, not compiled, not run on hardware
//
#include <hip/hip_runtime.h>

typedef _Float16 v16h __attribute__((ext_vector_type(16)));
typedef _Float16 v8h  __attribute__((ext_vector_type(8)));
typedef _Float16 v2h  __attribute__((ext_vector_type(2)));
typedef float    v8f  __attribute__((ext_vector_type(8)));

#define B_   16
#define S_   2048
#define D_   64
#define QT   128     // q rows per block
#define KT   64      // kv rows per tile
#define NW   8       // waves per block

__device__ __forceinline__ v8h cvt8(float4 a, float4 b, float s) {
  v8h h;
  h[0]=(_Float16)(a.x*s); h[1]=(_Float16)(a.y*s); h[2]=(_Float16)(a.z*s); h[3]=(_Float16)(a.w*s);
  h[4]=(_Float16)(b.x*s); h[5]=(_Float16)(b.y*s); h[6]=(_Float16)(b.z*s); h[7]=(_Float16)(b.w*s);
  return h;
}

__device__ __forceinline__ v16h cat(v8h lo, v8h hi) {
  v16h r;
#pragma unroll
  for (int i = 0; i < 8; ++i) { r[i] = lo[i]; r[i+8] = hi[i]; }
  return r;
}

template<int CTRL>
__device__ __forceinline__ float dppmax(float x) {
  int y = __builtin_amdgcn_mov_dpp(__float_as_int(x), CTRL, 0xF, 0xF, true);
  return fmaxf(x, __int_as_float(y));
}
__device__ __forceinline__ float rowmax16(float x) {
  x = dppmax<0xB1>(x);   // quad_perm [1,0,3,2]  : xor 1
  x = dppmax<0x4E>(x);   // quad_perm [2,3,0,1]  : xor 2
  x = dppmax<0x141>(x);  // row_half_mirror      : xor 4
  x = dppmax<0x140>(x);  // row_mirror           : xor 8
  return x;
}

// raw v_exp_f32 (no denormal fixup): args are always <= 0, underflow -> 0 is desired
__device__ __forceinline__ float fexp2(float x) { return __builtin_amdgcn_exp2f(x); }

// CDNA5 async copy: GLOBAL -> LDS, 16B per lane, tracked by ASYNCcnt
__device__ __forceinline__ void async_cp16(unsigned lds_off, const void* gaddr) {
  asm volatile("global_load_async_to_lds_b128 %0, %1, off"
               :: "v"(lds_off), "v"(gaddr) : "memory");
}
__device__ __forceinline__ void wait_async0() {
  asm volatile("s_wait_asynccnt 0" ::: "memory");
}

// ---------------- pre-pass: K -> f16 row-major, V -> f16 transposed+key-permuted ----------------
__global__ __launch_bounds__(256)
void prep_kv_f16(const float* __restrict__ K, const float* __restrict__ V,
                 _Float16* __restrict__ Kh, _Float16* __restrict__ VhT)
{
  alignas(16) __shared__ _Float16 Vsh[64 * 64];   // 8 KB
  const int b  = blockIdx.x / (S_ / 64);
  const int s0 = (blockIdx.x % (S_ / 64)) * 64;
  const int t  = threadIdx.x;
  const int r  = t >> 2;            // 0..63
  const int c  = (t & 3) * 16;      // 0,16,32,48

  const float* kp = K + ((size_t)(b * S_ + s0 + r)) * D_ + c;
  float4 k0 = *(const float4*)(kp+0), k1 = *(const float4*)(kp+4);
  float4 k2 = *(const float4*)(kp+8), k3 = *(const float4*)(kp+12);
  _Float16* ko = Kh + ((size_t)(b * S_ + s0 + r)) * D_ + c;
  *(v8h*)(ko    ) = cvt8(k0, k1, 1.0f);
  *(v8h*)(ko + 8) = cvt8(k2, k3, 1.0f);

  const float* vp = V + ((size_t)(b * S_ + s0 + r)) * D_ + c;
  float4 u0 = *(const float4*)(vp+0), u1 = *(const float4*)(vp+4);
  float4 u2 = *(const float4*)(vp+8), u3 = *(const float4*)(vp+12);
  const int j   = r >> 4, lnk = r & 15;
  const int pr  = 2 * lnk + (j & 1) + 32 * (j >> 1);   // key -> position (matches P packing)
  *(v8h*)&Vsh[pr * 64 + c    ] = cvt8(u0, u1, 1.0f);
  *(v8h*)&Vsh[pr * 64 + c + 8] = cvt8(u2, u3, 1.0f);
  __syncthreads();

  v8h t0, t1;
#pragma unroll
  for (int i = 0; i < 8; ++i) {
    t0[i] = Vsh[(c + i) * 64 + r];
    t1[i] = Vsh[(c + 8 + i) * 64 + r];
  }
  _Float16* vo = VhT + ((size_t)(b * D_ + r)) * S_ + s0 + c;
  *(v8h*)(vo    ) = t0;
  *(v8h*)(vo + 8) = t1;
}

// ---------------- main flash-attention kernel ----------------
__global__ __launch_bounds__(256)
void fa_fwd_wmma_f16(const float* __restrict__ Q, const _Float16* __restrict__ Kh,
                     const _Float16* __restrict__ VhT, float* __restrict__ O)
{
  alignas(16) __shared__ _Float16 KVbuf[2][2][KT * D_];  // double-buffered K|V tiles, 32 KB
  alignas(16) __shared__ _Float16 Psh [NW * 16 * KT];    // per-wave P tiles, 16 KB

  const int tid  = threadIdx.x;
  const int lane = tid & 31;
  const int w    = tid >> 5;
  const int lg   = (lane & 16) ? 1 : 0;
  const int ln   = lane & 15;

  const int batch = blockIdx.x / (S_ / QT);
  const int qbase = (blockIdx.x % (S_ / QT)) * QT;
  const float qs = 0.125f * 1.44269504088896340736f;   // 1/sqrt(D) * log2(e)

  // Q strip -> two pre-scaled f16 A-fragments
  const int qrow = qbase + w * 16 + ln;
  const float* Qp = Q + ((size_t)(batch * S_ + qrow)) * D_;
  v16h aq[2];
#pragma unroll
  for (int f = 0; f < 2; ++f) {
    const int c1 = f * 32 + (lg ? 16 : 0);
    const int c2 = c1 + 8;
    float4 x0 = *(const float4*)(Qp + c1);
    float4 x1 = *(const float4*)(Qp + c1 + 4);
    float4 x2 = *(const float4*)(Qp + c2);
    float4 x3 = *(const float4*)(Qp + c2 + 4);
    aq[f] = cat(cvt8(x0, x1, qs), cvt8(x2, x3, qs));
  }

  // all-ones B fragment for the denominator WMMA; pin it in VGPRs so it is not
  // rematerialized from SGPRs in every iteration
  v16h ones;
#pragma unroll
  for (int i = 0; i < 16; ++i) ones[i] = (_Float16)1.0f;
  asm volatile("" : "+v"(ones));

  float rm[8];
  v8f o[5];
#pragma unroll
  for (int g = 0; g < 8; ++g) rm[g] = -3.0e38f;
#pragma unroll
  for (int t = 0; t < 5; ++t) { v8f z = {}; o[t] = z; }

  const unsigned ldsBase = (unsigned)(size_t)&KVbuf[0][0][0];
  const char* kgBase = (const char*)(Kh + ((size_t)batch * S_) * D_);
  const char* vgBase = (const char*)(VhT + (size_t)batch * D_ * S_);
  const int   vdr = tid >> 3;            // 0..31
  const unsigned vch = (tid & 7) * 16;   // 16B chunk in a 128B row

  _Float16* Pw = &Psh[w * 16 * KT];
  const int NT = S_ / KT;

  auto issue_tile = [&](int kv0, int bi) {
    const unsigned ldsK = ldsBase + (unsigned)(bi * 2) * (KT * D_ * 2);
    const unsigned ldsV = ldsK + KT * D_ * 2;
    const char* kg = kgBase + (size_t)kv0 * D_ * 2;       // contiguous 8 KB
#pragma unroll
    for (int i = 0; i < 2; ++i) {                          // 256 thr x 16B = 4 KB per issue
      const unsigned off = (unsigned)tid * 16 + (unsigned)i * 4096;
      async_cp16(ldsK + off, kg + off);
    }
    const char* vg = vgBase + (size_t)kv0 * 2;            // 64 rows x 128B, stride S_*2
#pragma unroll
    for (int i = 0; i < 2; ++i) {
      const int d = i * 32 + vdr;
      async_cp16(ldsV + (unsigned)d * 128 + vch, vg + (size_t)d * (S_ * 2) + vch);
    }
  };

  issue_tile(0, 0);

#pragma unroll 2
  for (int kt = 0; kt < NT; ++kt) {
    wait_async0();        // this wave's copies for tile kt done
    __syncthreads();      // everyone's copies done; prior buffer free for reuse
    const int bi = kt & 1;
    if (kt + 1 < NT) issue_tile(kt + 1, bi ^ 1);   // overlap next copy with compute

    const _Float16* Kb = &KVbuf[bi][0][0];
    const _Float16* Vb = &KVbuf[bi][1][0];

    // ---- scores: four 16x16 C tiles of Qs . K^T ----
    v8f s[4];
    const int d0 = lg ? 16 : 0;
#pragma unroll
    for (int j = 0; j < 4; ++j) {
      const int kr = j * 16 + ln;
      v16h bk0 = cat(*(const v8h*)&Kb[kr * D_ +      d0], *(const v8h*)&Kb[kr * D_ +      d0 + 8]);
      v16h bk1 = cat(*(const v8h*)&Kb[kr * D_ + 32 + d0], *(const v8h*)&Kb[kr * D_ + 32 + d0 + 8]);
      v8f z = {};
      z = __builtin_amdgcn_wmma_f32_16x16x32_f16(false, aq[0], false, bk0, (short)0, z, false, false);
      z = __builtin_amdgcn_wmma_f32_16x16x32_f16(false, aq[1], false, bk1, (short)0, z, false, false);
      s[j] = z;
    }

    // ---- online softmax ----
    // rm[] is replicated across the 16-lane group, so a per-lane test of the
    // 4-way local max against rm already detects any row-max growth: the full
    // DPP reduction + rescale runs only when __ballot says some row grew.
    float mnew[8];
    int inc = 0;
#pragma unroll
    for (int g = 0; g < 8; ++g) {
      mnew[g] = fmaxf(fmaxf(s[0][g], s[1][g]), fmaxf(s[2][g], s[3][g]));  // local 4-way max
      inc |= (mnew[g] > rm[g]) ? 1 : 0;
    }
    if (__ballot(inc)) {               // wave-uniform branch; EXEC stays all-ones
#pragma unroll
      for (int g = 0; g < 8; ++g) {
        const float mx = rowmax16(mnew[g]);
        const float mn = fmaxf(rm[g], mx);
        const float alpha = fexp2(rm[g] - mn);
        rm[g] = mn;
        mnew[g] = mn;
#pragma unroll
        for (int t = 0; t < 5; ++t) o[t][g] *= alpha;
      }
    } else {
#pragma unroll
      for (int g = 0; g < 8; ++g) mnew[g] = rm[g];
    }
#pragma unroll
    for (int g = 0; g < 8; ++g) {
      v2h q01, q23;
      q01[0] = (_Float16)fexp2(s[0][g] - mnew[g]);
      q01[1] = (_Float16)fexp2(s[1][g] - mnew[g]);
      q23[0] = (_Float16)fexp2(s[2][g] - mnew[g]);
      q23[1] = (_Float16)fexp2(s[3][g] - mnew[g]);
      const int row = g + lg * 8;
      *(v2h*)&Pw[row * KT + 2 * ln     ] = q01;   // keys j=0,1 -> pos 2ln, 2ln+1
      *(v2h*)&Pw[row * KT + 2 * ln + 32] = q23;   // keys j=2,3 -> pos 32+2ln, 33+2ln
    }
    __syncthreads();   // P visible across lanes

    // ---- O += P . V (positions permuted consistently), denominator via ones-WMMA ----
    {
      const int pc = lg ? 16 : 0;
      const int ks = lg ? 16 : 0;
      v16h ap0 = cat(*(const v8h*)&Pw[ln * KT +      pc], *(const v8h*)&Pw[ln * KT +      pc + 8]);
      v16h ap1 = cat(*(const v8h*)&Pw[ln * KT + 32 + pc], *(const v8h*)&Pw[ln * KT + 32 + pc + 8]);
#pragma unroll
      for (int t = 0; t < 4; ++t) {
        const int dr = t * 16 + ln;
        v16h bv0 = cat(*(const v8h*)&Vb[dr * KT +      ks], *(const v8h*)&Vb[dr * KT +      ks + 8]);
        v16h bv1 = cat(*(const v8h*)&Vb[dr * KT + 32 + ks], *(const v8h*)&Vb[dr * KT + 32 + ks + 8]);
        o[t] = __builtin_amdgcn_wmma_f32_16x16x32_f16(false, ap0, false, bv0, (short)0, o[t], false, false);
        o[t] = __builtin_amdgcn_wmma_f32_16x16x32_f16(false, ap1, false, bv1, (short)0, o[t], false, false);
      }
      o[4] = __builtin_amdgcn_wmma_f32_16x16x32_f16(false, ap0, false, ones, (short)0, o[4], false, false);
      o[4] = __builtin_amdgcn_wmma_f32_16x16x32_f16(false, ap1, false, ones, (short)0, o[4], false, false);
    }
  }

  // ---- normalize by WMMA-accumulated row sums (raw v_rcp_f32), store f32 ----
#pragma unroll
  for (int g = 0; g < 8; ++g) {
    const int row = qbase + w * 16 + g + lg * 8;
    const float inv = __builtin_amdgcn_rcpf(o[4][g]);
#pragma unroll
    for (int t = 0; t < 4; ++t)
      O[((size_t)(batch * S_ + row)) * D_ + t * 16 + ln] = o[t][g] * inv;
  }
}

extern "C" void kernel_launch(void* const* d_in, const int* in_sizes, int n_in,
                              void* d_out, int out_size, void* d_ws, size_t ws_size,
                              hipStream_t stream) {
  (void)in_sizes; (void)n_in; (void)out_size; (void)ws_size;
  const float* Q = (const float*)d_in[0];
  const float* K = (const float*)d_in[1];
  const float* V = (const float*)d_in[2];
  float* O = (float*)d_out;

  _Float16* Kh  = (_Float16*)d_ws;                       // 4 MB
  _Float16* VhT = Kh + (size_t)B_ * S_ * D_;             // 4 MB

  hipLaunchKernelGGL(prep_kv_f16, dim3(B_ * (S_ / 64)), dim3(256), 0, stream, K, V, Kh, VhT);
  hipLaunchKernelGGL(fa_fwd_wmma_f16, dim3(B_ * (S_ / QT)), dim3(NW * 32), 0, stream, Q, Kh, VhT, O);
}
